// Base_LZD_Encoder_86354612453845
// MI455X (gfx1250) — matrix operand order repaired and checked by reading, not compiled, not run on hardware
//
#include <hip/hip_runtime.h>

#define N_SEQ   8
#define MAX_LEN 4096
#define H_DIM   1024
#define K_DIM   2048   // 2*H_DIM
#define E_GRP   16384
#define L_GRP   12

typedef __attribute__((ext_vector_type(16))) __bf16 v16bf;
typedef __attribute__((ext_vector_type(8)))  float  v8f;
typedef __attribute__((ext_vector_type(4)))  float  f32x4;

union AFrag { v16bf v; __bf16 e[16]; };
union CFrag { v8f   v; float  e[8]; };

struct ARaw { f32x4 p0, p1, p2, p3; };     // one 16x32 A fragment, raw fp32
struct BFr  { v16bf b0, b1, b2, b3; };     // four 32x16 B fragments (N-tiles)

__device__ __forceinline__ ARaw load_a(const float* __restrict__ row, int c0) {
    ARaw r;
    r.p0 = *(const f32x4*)(row + c0);
    r.p1 = *(const f32x4*)(row + c0 + 4);
    r.p2 = *(const f32x4*)(row + c0 + 8);
    r.p3 = *(const f32x4*)(row + c0 + 12);
    return r;
}

__device__ __forceinline__ v16bf cvt_a(const ARaw& r) {
    AFrag a;
#pragma unroll
    for (int j = 0; j < 4; ++j) {
        a.e[j]      = (__bf16)r.p0[j];
        a.e[4 + j]  = (__bf16)r.p1[j];
        a.e[8 + j]  = (__bf16)r.p2[j];
        a.e[12 + j] = (__bf16)r.p3[j];
    }
    return a.v;
}

__device__ __forceinline__ BFr load_b(const __bf16* __restrict__ wp) {
    BFr b;
    b.b0 = *(const v16bf*)(wp);
    b.b1 = *(const v16bf*)(wp + 16 * K_DIM);
    b.b2 = *(const v16bf*)(wp + 32 * K_DIM);
    b.b3 = *(const v16bf*)(wp + 48 * K_DIM);
    return b;
}

// ---------------------------------------------------------------- zero fill
__global__ void __launch_bounds__(256) zero_f4(float* __restrict__ p, int n4) {
    int i = blockIdx.x * 256 + threadIdx.x;
    if (i < n4) {
        f32x4 z = {0.f, 0.f, 0.f, 0.f};
        ((f32x4*)p)[i] = z;
    }
}

// ------------------------------------------------- W[k][n] -> Wt[n][k] bf16
__global__ void __launch_bounds__(256) cvt_w(const float* __restrict__ W,
                                             __bf16* __restrict__ Wt) {
    long t = (long)blockIdx.x * 256 + threadIdx.x;   // over 1024*2048
    int n = (int)(t >> 11);
    int k = (int)(t & 2047);
    Wt[t] = (__bf16)W[(long)k * H_DIM + n];
}

// ------------------------------------------------------ leaf char scatter
__global__ void __launch_bounds__(256) char_scatter(const int* __restrict__ i_seq,
                                                    const int* __restrict__ i_pos,
                                                    const int* __restrict__ ids,
                                                    const float* __restrict__ emb,
                                                    float* __restrict__ h) {
    int e = blockIdx.x;
    int c = threadIdx.x * 4;
    const float* src = emb + (long)ids[e] * H_DIM + c;
    float* dst = h + ((long)i_seq[e] * MAX_LEN + i_pos[e]) * H_DIM + c;
    f32x4 v = *(const f32x4*)src;
    atomicAdd(dst + 0, v[0]);
    atomicAdd(dst + 1, v[1]);
    atomicAdd(dst + 2, v[2]);
    atomicAdd(dst + 3, v[3]);
}

// --------------------------------------------------------- compose GEMM
// out[e, n] = tanh( sum_k concat(h[rowL], h[rowR])[k] * W[k][n] + b[n] )
// M = E_GRP, N = H_DIM, K = K_DIM.  bf16 WMMA, fp32 accumulate.
// Block: 256 threads = 8 waves in a 2(M) x 4(N) grid.
// Block tile: M=32, N=256.  Wave tile: 16x64 (4 WMMA C frags).
// Two-stage structural ping-pong (no register rotation), #pragma unroll 1
// keeps the steady state a tight loop with loads a full stage ahead of use.
__global__ void __launch_bounds__(256) compose_gemm(const float*  __restrict__ h,
                                                    const __bf16* __restrict__ Wt,
                                                    const float*  __restrict__ bias,
                                                    const int*    __restrict__ i_seq,
                                                    const int*    __restrict__ i_first,
                                                    const int*    __restrict__ i_second,
                                                    float*        __restrict__ outbuf) {
    const int lane  = threadIdx.x & 31;
    const int wave  = threadIdx.x >> 5;
    const int r     = lane & 15;          // A: row M; B/C: column N
    const int hi    = lane >> 4;          // lane-group selector
    const int mrow  = wave >> 2;          // 0..1 : which 16-row strip
    const int mbase = blockIdx.x * 32 + mrow * 16;
    const int nbase = blockIdx.y * 256 + (wave & 3) * 64;

    const int e    = mbase + r;
    const long seq = i_seq[e];
    const float* rowL = h + (seq * MAX_LEN + i_first[e])  * (long)H_DIM;
    const float* rowR = h + (seq * MAX_LEN + i_second[e]) * (long)H_DIM;

    const int colA = hi ? 16 : 0;         // lane-group K offset inside chunk
    const __bf16* wp0 = Wt + (long)(nbase + r) * K_DIM + hi * 16;

    v8f c0 = {}, c1 = {}, c2 = {}, c3 = {};

#define WMMA4(AF, B)                                                                                    \
    c0 = __builtin_amdgcn_wmma_f32_16x16x32_bf16(false, AF, false, (B).b0, (short)0, c0, false, false); \
    c1 = __builtin_amdgcn_wmma_f32_16x16x32_bf16(false, AF, false, (B).b1, (short)0, c1, false, false); \
    c2 = __builtin_amdgcn_wmma_f32_16x16x32_bf16(false, AF, false, (B).b2, (short)0, c2, false, false); \
    c3 = __builtin_amdgcn_wmma_f32_16x16x32_bf16(false, AF, false, (B).b3, (short)0, c3, false, false)

    ARaw a0, a1;
    BFr  b0, b1;

    // ---- prologue: chunk 0 (left half)
    a0 = load_a(rowL, colA);
    b0 = load_b(wp0);

    // ---- left half steady state: chunks 0..29 computed, pairs of 2
#pragma unroll 1
    for (int k = 0; k < 30 * 32; k += 64) {
        __builtin_prefetch(wp0 + k + 128, 0, 1);           // B stream ahead
        a1 = load_a(rowL, k + 32 + colA);
        b1 = load_b(wp0 + k + 32);
        { v16bf af = cvt_a(a0); WMMA4(af, b0); }           // chunk k/32
        a0 = load_a(rowL, k + 64 + colA);
        b0 = load_b(wp0 + k + 64);
        { v16bf af = cvt_a(a1); WMMA4(af, b1); }           // chunk k/32+1
    }

    // ---- boundary: chunks 30, 31; preload right-half chunk 32
    a1 = load_a(rowL, 31 * 32 + colA);
    b1 = load_b(wp0 + 31 * 32);
    { v16bf af = cvt_a(a0); WMMA4(af, b0); }               // chunk 30
    a0 = load_a(rowR, colA);
    b0 = load_b(wp0 + H_DIM);
    { v16bf af = cvt_a(a1); WMMA4(af, b1); }               // chunk 31

    // ---- right half steady state: chunks 32..61
#pragma unroll 1
    for (int k = 0; k < 30 * 32; k += 64) {
        __builtin_prefetch(wp0 + H_DIM + k + 128, 0, 1);
        a1 = load_a(rowR, k + 32 + colA);
        b1 = load_b(wp0 + H_DIM + k + 32);
        { v16bf af = cvt_a(a0); WMMA4(af, b0); }
        a0 = load_a(rowR, k + 64 + colA);
        b0 = load_b(wp0 + H_DIM + k + 64);
        { v16bf af = cvt_a(a1); WMMA4(af, b1); }
    }

    // ---- tail: chunks 62, 63
    a1 = load_a(rowR, 31 * 32 + colA);
    b1 = load_b(wp0 + H_DIM + 31 * 32);
    { v16bf af = cvt_a(a0); WMMA4(af, b0); }               // chunk 62
    { v16bf af = cvt_a(a1); WMMA4(af, b1); }               // chunk 63
#undef WMMA4

    // ---- epilogue: bias + tanh, store out[e][n]
    v8f acc[4] = {c0, c1, c2, c3};
#pragma unroll
    for (int t = 0; t < 4; ++t) {
        const int n = nbase + t * 16 + r;
        const float bv = bias[n];
        CFrag cf; cf.v = acc[t];
#pragma unroll
        for (int j = 0; j < 8; ++j) {
            const int m = hi * 8 + j;          // C layout: VGPR j <-> M = 8*hi + j
            outbuf[(long)(mbase + m) * H_DIM + n] = tanhf(cf.e[j] + bv);
        }
    }
}

// ------------------------------------------------------ compose scatter-add
__global__ void __launch_bounds__(256) compose_scatter(const float* __restrict__ outbuf,
                                                       const int* __restrict__ i_seq,
                                                       const int* __restrict__ i_pos,
                                                       float* __restrict__ h) {
    int e = blockIdx.x;
    int c = threadIdx.x * 4;
    f32x4 v = *(const f32x4*)(outbuf + (long)e * H_DIM + c);
    float* dst = h + ((long)i_seq[e] * MAX_LEN + i_pos[e]) * H_DIM + c;
    atomicAdd(dst + 0, v[0]);
    atomicAdd(dst + 1, v[1]);
    atomicAdd(dst + 2, v[2]);
    atomicAdd(dst + 3, v[3]);
}

// ------------------------------------------------------------- h[:,0] = 0
__global__ void __launch_bounds__(256) zero_first_pos(float* __restrict__ h) {
    int s = blockIdx.x;           // N_SEQ blocks
    int c = threadIdx.x * 4;
    f32x4 z = {0.f, 0.f, 0.f, 0.f};
    *(f32x4*)(h + (long)s * MAX_LEN * H_DIM + c) = z;
}

extern "C" void kernel_launch(void* const* d_in, const int* in_sizes, int n_in,
                              void* d_out, int out_size, void* d_ws, size_t ws_size,
                              hipStream_t stream) {
    const int*   char_i_seq   = (const int*)d_in[0];
    const int*   char_i_pos   = (const int*)d_in[1];
    const int*   char_ids     = (const int*)d_in[2];
    const int*   grp_i_seq    = (const int*)d_in[3];
    const int*   grp_i_first  = (const int*)d_in[4];
    const int*   grp_i_second = (const int*)d_in[5];
    const int*   grp_i_pos    = (const int*)d_in[6];
    const float* emb_char = (const float*)d_in[7];
    const float* W        = (const float*)d_in[8];
    const float* b        = (const float*)d_in[9];

    float* h = (float*)d_out;                         // [8, 4096, 1024] fp32

    // Workspace layout: Wt bf16 (4 MB) | outbuf fp32 (64 MB)
    __bf16* Wt    = (__bf16*)d_ws;
    float* outbuf = (float*)((char*)d_ws + (size_t)K_DIM * H_DIM * sizeof(__bf16));

    const int n_char = in_sizes[0];

    // 1) h = 0
    const int n4 = N_SEQ * MAX_LEN * H_DIM / 4;
    zero_f4<<<(n4 + 255) / 256, 256, 0, stream>>>(h, n4);

    // 2) Wt[n][k] = bf16(W[k][n])   (4 MB, stays in L2 across all 12 levels)
    cvt_w<<<(K_DIM * H_DIM) / 256, 256, 0, stream>>>(W, Wt);

    // 3) leaf characters: h[seq,pos] += emb[char]
    char_scatter<<<n_char, 256, 0, stream>>>(char_i_seq, char_i_pos, char_ids, emb_char, h);

    // 4) 12 sequential compose levels. Kernel split (GEMM -> scatter) gives the
    //    gather-before-scatter ordering the reference scan requires.
    for (int lev = 0; lev < L_GRP; ++lev) {
        const int* gs  = grp_i_seq    + (long)lev * E_GRP;
        const int* gf  = grp_i_first  + (long)lev * E_GRP;
        const int* gs2 = grp_i_second + (long)lev * E_GRP;
        const int* gp  = grp_i_pos    + (long)lev * E_GRP;

        dim3 grid(E_GRP / 32, H_DIM / 256);           // 512 x 4 blocks, 256 thr
        compose_gemm<<<grid, 256, 0, stream>>>(h, Wt, b, gs, gf, gs2, outbuf);
        compose_scatter<<<E_GRP, 256, 0, stream>>>(outbuf, gs, gp, h);
    }

    // 5) h[:, 0, :] = 0
    zero_first_pos<<<N_SEQ, 256, 0, stream>>>(h);
}
